// PopulationODE_61297773248810
// MI455X (gfx1250) — matrix operand order repaired and checked by reading, not compiled, not run on hardware
//
#include <hip/hip_runtime.h>
#include <hip/hip_bf16.h>

typedef __attribute__((ext_vector_type(16))) _Float16 v16h;
typedef __attribute__((ext_vector_type(8)))  float    v8f;

#define T_STEPS 1024
#define B_TRAJ  32768
#define N_STATE 4
#define HID     32
#define WAVES_PER_BLOCK 4

// Each wave integrates 16 trajectories through all 1023 Euler steps.
// Layer GEMMs run on v_wmma_f32_16x16x32_f16; weights live in registers in
// B-operand layout; inter-layer transposes go through per-wave LDS tiles.
__global__ __launch_bounds__(WAVES_PER_BLOCK * 32)
void ode_scan_wmma(const float* __restrict__ s_grid,
                   const float* __restrict__ y0,
                   const float* __restrict__ W1, const float* __restrict__ b1,
                   const float* __restrict__ W2, const float* __restrict__ b2,
                   const float* __restrict__ W3, const float* __restrict__ b3,
                   float* __restrict__ out)
{
    __shared__ __align__(16) _Float16 hstage[WAVES_PER_BLOCK][16][HID];     // 4 KB
    __shared__ __align__(16) float    ystage[WAVES_PER_BLOCK][16][N_STATE]; // 1 KB

    const int  lane    = threadIdx.x & 31;
    const int  wv      = threadIdx.x >> 5;
    const int  gw      = blockIdx.x * WAVES_PER_BLOCK + wv;
    const int  rowbase = gw * 16;          // first trajectory of this wave's tile
    const int  nlo     = lane & 15;        // column index inside a 16-wide tile
    const bool hi      = lane >= 16;
    const int  kbase   = hi ? 16 : 0;      // B-layout K base for this half-wave
    const int  mofs    = hi ? 8 : 0;       // D-layout row base for this half-wave

    // ---- Load weights once into WMMA B-operand layout (lane=N, half i = K=kbase+i) ----
    v16h bW1[2], bW2[2], bW3;
    #pragma unroll
    for (int t = 0; t < 2; ++t) {
        const int n = t * 16 + nlo;
        #pragma unroll
        for (int i = 0; i < 16; ++i) {
            const int k = kbase + i;
            bW1[t][i] = (k < N_STATE) ? (_Float16)W1[k * HID + n] : (_Float16)0.0f;
            bW2[t][i] = (_Float16)W2[k * HID + n];
        }
    }
    #pragma unroll
    for (int i = 0; i < 16; ++i) {
        const int k = kbase + i;
        bW3[i] = (nlo < N_STATE) ? (_Float16)W3[k * N_STATE + nlo] : (_Float16)0.0f;
    }
    const float b1a = b1[nlo], b1b = b1[16 + nlo];
    const float b2a = b2[nlo], b2b = b2[16 + nlo];
    const float b3v = (nlo < N_STATE) ? b3[nlo] : 0.0f;

    // ---- Initial state in D-layout f32 registers: lane holds column j=nlo, rows r+mofs ----
    float yD[8];
    #pragma unroll
    for (int r = 0; r < 8; ++r) {
        const int m = r + mofs;
        yD[r] = (nlo < N_STATE) ? y0[(size_t)(rowbase + m) * N_STATE + nlo] : 0.0f;
    }

    // Publish y_0 into the f32 staging tile; coalesced store of out[0] (256 B per wave).
    if (nlo < N_STATE) {
        #pragma unroll
        for (int r = 0; r < 8; ++r) ystage[wv][r + mofs][nlo] = yD[r];
    }
    {
        const float* yflat = &ystage[wv][0][0];
        float* dst = out + (size_t)rowbase * N_STATE;
        *(float2*)(dst + lane * 2) = *(const float2*)(yflat + lane * 2);
    }

    for (int t = 0; t < T_STEPS - 1; ++t) {
        const float hstep = s_grid[t + 1] - s_grid[t];   // uniform scalar per step

        // ---- A operand for layer 1: row m=nlo (low lanes), K 0..3 = y, rest zero ----
        v16h Ay = {};
        if (!hi) {
            #pragma unroll
            for (int j = 0; j < N_STATE; ++j)
                Ay[j] = (_Float16)ystage[wv][nlo][j];
        }

        v8f cz = {};
        // ---- Layer 1: [16x32(pad)] x [32x32] via two N-halves ----
        v8f d0 = __builtin_amdgcn_wmma_f32_16x16x32_f16(false, Ay, false, bW1[0], (short)0, cz, false, false);
        v8f d1 = __builtin_amdgcn_wmma_f32_16x16x32_f16(false, Ay, false, bW1[1], (short)0, cz, false, false);
        #pragma unroll
        for (int r = 0; r < 8; ++r) {
            float x0 = d0[r] + b1a;  x0 = x0 > 0.f ? x0 : __expf(x0) - 1.f;  // ELU
            float x1 = d1[r] + b1b;  x1 = x1 > 0.f ? x1 : __expf(x1) - 1.f;
            hstage[wv][r + mofs][nlo]      = (_Float16)x0;   // D-layout -> row-major LDS tile
            hstage[wv][r + mofs][nlo + 16] = (_Float16)x1;
        }
        // Re-load in A-layout: row m=nlo; low lanes K 0..15, high lanes K 16..31
        v16h Ah;
        {
            const _Float16* rowp = &hstage[wv][nlo][0];
            const int off = hi ? 16 : 0;
            #pragma unroll
            for (int i = 0; i < 8; ++i) { Ah[i] = rowp[off + i]; Ah[8 + i] = rowp[off + 8 + i]; }
        }

        // ---- Layer 2: [16x32] x [32x32] ----
        d0 = __builtin_amdgcn_wmma_f32_16x16x32_f16(false, Ah, false, bW2[0], (short)0, cz, false, false);
        d1 = __builtin_amdgcn_wmma_f32_16x16x32_f16(false, Ah, false, bW2[1], (short)0, cz, false, false);
        #pragma unroll
        for (int r = 0; r < 8; ++r) {
            float x0 = d0[r] + b2a;  x0 = x0 > 0.f ? x0 : __expf(x0) - 1.f;
            float x1 = d1[r] + b2b;  x1 = x1 > 0.f ? x1 : __expf(x1) - 1.f;
            hstage[wv][r + mofs][nlo]      = (_Float16)x0;
            hstage[wv][r + mofs][nlo + 16] = (_Float16)x1;
        }
        v16h Ah2;
        {
            const _Float16* rowp = &hstage[wv][nlo][0];
            const int off = hi ? 16 : 0;
            #pragma unroll
            for (int i = 0; i < 8; ++i) { Ah2[i] = rowp[off + i]; Ah2[8 + i] = rowp[off + 8 + i]; }
        }

        // ---- Layer 3: [16x32] x [32x4(pad to 16)] -> D-layout matches yD directly ----
        v8f dy = __builtin_amdgcn_wmma_f32_16x16x32_f16(false, Ah2, false, bW3, (short)0, cz, false, false);
        #pragma unroll
        for (int r = 0; r < 8; ++r) yD[r] += hstep * (dy[r] + b3v);   // f32 Euler update

        // ---- Publish y_{t+1}; coalesced 256 B store of out[t+1] ----
        if (nlo < N_STATE) {
            #pragma unroll
            for (int r = 0; r < 8; ++r) ystage[wv][r + mofs][nlo] = yD[r];
        }
        const float* yflat = &ystage[wv][0][0];
        float* dst = out + ((size_t)(t + 1) * B_TRAJ + rowbase) * N_STATE;
        *(float2*)(dst + lane * 2) = *(const float2*)(yflat + lane * 2);
    }
}

extern "C" void kernel_launch(void* const* d_in, const int* in_sizes, int n_in,
                              void* d_out, int out_size, void* d_ws, size_t ws_size,
                              hipStream_t stream) {
    (void)in_sizes; (void)n_in; (void)d_ws; (void)ws_size; (void)out_size;
    const float* s_grid = (const float*)d_in[0];
    const float* y0     = (const float*)d_in[1];
    const float* W1     = (const float*)d_in[2];
    const float* b1     = (const float*)d_in[3];
    const float* W2     = (const float*)d_in[4];
    const float* b2     = (const float*)d_in[5];
    const float* W3     = (const float*)d_in[6];
    const float* b3     = (const float*)d_in[7];
    float* out = (float*)d_out;

    const int total_waves = B_TRAJ / 16;                 // 2048 independent tiles
    const int blocks      = total_waves / WAVES_PER_BLOCK; // 512 blocks x 128 threads
    ode_scan_wmma<<<blocks, WAVES_PER_BLOCK * 32, 0, stream>>>(
        s_grid, y0, W1, b1, W2, b2, W3, b3, out);
}
